// IsomorphicCell_14353780703961
// MI455X (gfx1250) — matrix operand order repaired and checked by reading, not compiled, not run on hardware
//
#include <hip/hip_runtime.h>
#include <hip/hip_bf16.h>
#include <math.h>

// ---------------------------------------------------------------------------
// CDNA5 (gfx1250) implementation of the 2-layer hetero-GIN + MLP head.
// v_wmma_f32_16x16x32_f16 (wave32), one wave per 16-row M-strip with A-frag
// reuse across all n-tiles. Weights are PRE-PACKED into WMMA B-fragment order
// so each B fragment is one contiguous 32B v16h load per lane (b128 pairs).
// A-tile loads use 8-float vector loads matching the ISA A fragment runs.
// Scatter-add uses global_atomic_add_f32 (dest is L2-resident, 12.8MB<<192MB).
// ---------------------------------------------------------------------------

typedef __attribute__((ext_vector_type(16))) _Float16 v16h;
typedef __attribute__((ext_vector_type(8)))  float    v8f;
typedef __attribute__((ext_vector_type(8)))  float    f32x8;

#define NFEAT 64
#define GH    128

__device__ __forceinline__ float gelu_f(float x) {
    // tanh-approx GELU (jax.nn.gelu default)
    float x3 = x * x * x;
    float t  = tanhf(0.7978845608028654f * (x + 0.044715f * x3));
    return 0.5f * x * (1.0f + t);
}

// ------------------------------- utilities ---------------------------------

__global__ void zero_kernel(float* __restrict__ p, int n) {
    int i = blockIdx.x * blockDim.x + threadIdx.x;
    if (i < n) p[i] = 0.0f;
}

// Pack W[K][Nout] (f32, row-major) into WMMA B-fragment order:
//   dst[((nt*(K/32) + kc)*32 + lane)*16 + i] = W[k][n]
//   with k = kc*32 + (lane>>4)*8 + (i&7) + (i>>3)*16,  n = nt*16 + (lane&15)
// so a wave's B fragment is 32 lanes x 32 contiguous bytes (coalesced b128s).
__global__ void pack_b_kernel(const float* __restrict__ W,
                              _Float16* __restrict__ dst, int K, int Nout) {
    int idx = blockIdx.x * blockDim.x + threadIdx.x;
    if (idx >= K * Nout) return;
    int i    = idx & 15;
    int lane = (idx >> 4) & 31;
    int fc   = idx >> 9;                // fragment index
    int kchunks = K >> 5;
    int nt = fc / kchunks;
    int kc = fc - nt * kchunks;
    int k  = kc * 32 + (lane >> 4) * 8 + (i & 7) + (i >> 3) * 16;
    int n  = nt * 16 + (lane & 15);
    dst[idx] = (_Float16)W[k * Nout + n];
}

// --------------------------- edge scatter-add ------------------------------
// agg[dst] += h[src], 4 features per thread, 16 threads per edge
// (16 consecutive threads cover one edge's 256 contiguous bytes).
__global__ void scatter_kernel(const float* __restrict__ h,
                               const int* __restrict__ src,
                               const int* __restrict__ dst,
                               float* __restrict__ agg, int total) {
    int t = blockIdx.x * blockDim.x + threadIdx.x;
    if (t >= total) return;
    int e = t >> 4;
    int c = (t & 15) * 4;
    int s = src[e];
    int d = dst[e];
    const float4 v = *(const float4*)(h + (long long)s * NFEAT + c);
    float* o = agg + (long long)d * NFEAT + c;
    atomicAdd(o + 0, v.x);
    atomicAdd(o + 1, v.y);
    atomicAdd(o + 2, v.z);
    atomicAdd(o + 3, v.w);
}

// ------------------------------- GEMM 1 ------------------------------------
// z[N,128] = ((1+eps)*h + agg)[N,64] @ W1[64,128] + b1.
// One wave per 16-row M-strip; A fragments built once, reused for 8 n-tiles.
__global__ void gemm1_kernel(const float* __restrict__ h,
                             const float* __restrict__ agg,
                             const float* __restrict__ epsp,
                             const v16h* __restrict__ Bf,   // packed [16][32]
                             const float* __restrict__ bias,    // [128]
                             float* __restrict__ z,             // [N][128]
                             int N) {
    const int lane = threadIdx.x & 31;
    const int wave = threadIdx.x >> 5;
    const int mt   = blockIdx.x * 4 + wave;
    if (mt * 16 >= N) return;                   // wave-uniform
    const int half = lane >> 4;
    const int l15  = lane & 15;
    const int m    = mt * 16 + l15;
    const float ope = 1.0f + epsp[0];

    // A (16x32 f16) ISA layout: per lane two 8-float runs per 32-K chunk:
    //   a[i<8]  <- k = base + half*8 + i,  a[i>=8] <- k = base + 16 + half*8 + i
    const float* hp = h + (long long)m * NFEAT;
    const float* gp = agg + (long long)m * NFEAT;
    v16h a0, a1;
#pragma unroll
    for (int run = 0; run < 4; run++) {         // runs at k = half*8 + run*16
        const int kb = half * 8 + run * 16;
        f32x8 hv = *(const f32x8*)(hp + kb);
        f32x8 gv = *(const f32x8*)(gp + kb);
#pragma unroll
        for (int i = 0; i < 8; i++) {
            _Float16 val = (_Float16)(ope * hv[i] + gv[i]);
            if (run == 0)      a0[i]     = val;
            else if (run == 1) a0[i + 8] = val;
            else if (run == 2) a1[i]     = val;
            else               a1[i + 8] = val;
        }
    }

#pragma unroll
    for (int nt = 0; nt < GH / 16; nt++) {
        const int ncol = nt * 16 + l15;
        v16h b0 = Bf[(nt * 2 + 0) * 32 + lane];  // one 32B load per lane
        v16h b1 = Bf[(nt * 2 + 1) * 32 + lane];
        v8f c = {};
        c = __builtin_amdgcn_wmma_f32_16x16x32_f16(false, a0, false, b0,
                                                   (short)0, c, false, false);
        c = __builtin_amdgcn_wmma_f32_16x16x32_f16(false, a1, false, b1,
                                                   (short)0, c, false, false);
        const float bv = bias[ncol];
#pragma unroll
        for (int r = 0; r < 8; r++) {
            int mo = mt * 16 + r + half * 8;    // D layout: m = r + half*8
            z[mo * GH + ncol] = c[r] + bv;
        }
    }
}

// --------------------- BatchNorm column statistics -------------------------
// One block per column j of z[N,128]: stats[j]=mean, stats[128+j]=rsqrt(var+eps)
__global__ void colstats_kernel(const float* __restrict__ z,
                                float* __restrict__ stats, int N) {
    __shared__ float ss[256];
    __shared__ float sq[256];
    const int j = blockIdx.x;
    float s = 0.0f, q = 0.0f;
    for (int i = threadIdx.x; i < N; i += blockDim.x) {
        float v = z[(long long)i * GH + j];
        s += v;
        q += v * v;
    }
    ss[threadIdx.x] = s;
    sq[threadIdx.x] = q;
    __syncthreads();
    for (int st = 128; st > 0; st >>= 1) {
        if ((int)threadIdx.x < st) {
            ss[threadIdx.x] += ss[threadIdx.x + st];
            sq[threadIdx.x] += sq[threadIdx.x + st];
        }
        __syncthreads();
    }
    if (threadIdx.x == 0) {
        float m   = ss[0] / (float)N;
        float var = sq[0] / (float)N - m * m;
        stats[j]      = m;
        stats[GH + j] = rsqrtf(var + 1e-5f);
    }
}

// ------------------------------- GEMM 2 ------------------------------------
// out[N,64] (+)= gelu(BN(z)) @ W2[128,64] + b2.
// One wave per 16-row M-strip; gelu/BN computed ONCE per A element,
// A fragments (4 x v16h) reused across all 4 n-tiles.
__global__ void gemm2_kernel(const float* __restrict__ z,
                             const float* __restrict__ stats,  // mean|inv
                             const float* __restrict__ g1,
                             const float* __restrict__ bt1,
                             const v16h* __restrict__ Bf,      // packed [16][32]
                             const float* __restrict__ bias,   // [64]
                             float* __restrict__ outsum,       // [N][64]
                             int N, int accum) {
    const int lane = threadIdx.x & 31;
    const int wave = threadIdx.x >> 5;
    const int mt   = blockIdx.x * 4 + wave;
    if (mt * 16 >= N) return;
    const int half = lane >> 4;
    const int l15  = lane & 15;
    const int m    = mt * 16 + l15;

    const float* zp = z + (long long)m * GH;
    v16h a[4];
#pragma unroll
    for (int run = 0; run < 8; run++) {         // 8 runs of 8 floats cover K=128
        const int kb = run * 16 + half * 8;
        f32x8 zv = *(const f32x8*)(zp + kb);
#pragma unroll
        for (int i = 0; i < 8; i++) {
            int k = kb + i;
            float t = (zv[i] - stats[k]) * stats[GH + k] * g1[k] + bt1[k];
            a[run >> 1][(run & 1) * 8 + i] = (_Float16)gelu_f(t);
        }
    }

#pragma unroll
    for (int nt = 0; nt < NFEAT / 16; nt++) {
        const int ncol = nt * 16 + l15;
        v8f c = {};
#pragma unroll
        for (int kc = 0; kc < 4; kc++) {
            v16h b = Bf[(nt * 4 + kc) * 32 + lane];
            c = __builtin_amdgcn_wmma_f32_16x16x32_f16(false, a[kc], false, b,
                                                       (short)0, c, false, false);
        }
        const float bv = bias[ncol];
#pragma unroll
        for (int r = 0; r < 8; r++) {
            int mo = mt * 16 + r + half * 8;
            float v = c[r] + bv;
            if (accum) v += outsum[mo * NFEAT + ncol];
            outsum[mo * NFEAT + ncol] = v;
        }
    }
}

// -------------------------- plain 64x64 GEMM (head) ------------------------
__global__ void gemm_plain_kernel(const float* __restrict__ in,   // [N][64]
                                  const v16h* __restrict__ Bf,    // packed [8][32]
                                  const float* __restrict__ bias, // [64]
                                  float* __restrict__ out,        // [N][64]
                                  int N, int geluOut) {
    const int lane = threadIdx.x & 31;
    const int wave = threadIdx.x >> 5;
    const int mt   = blockIdx.x * 4 + wave;
    if (mt * 16 >= N) return;
    const int half = lane >> 4;
    const int l15  = lane & 15;
    const int m    = mt * 16 + l15;

    const float* ip = in + (long long)m * NFEAT;
    v16h a0, a1;
#pragma unroll
    for (int run = 0; run < 4; run++) {
        const int kb = half * 8 + run * 16;
        f32x8 iv = *(const f32x8*)(ip + kb);
#pragma unroll
        for (int i = 0; i < 8; i++) {
            _Float16 val = (_Float16)iv[i];
            if (run == 0)      a0[i]     = val;
            else if (run == 1) a0[i + 8] = val;
            else if (run == 2) a1[i]     = val;
            else               a1[i + 8] = val;
        }
    }

#pragma unroll
    for (int nt = 0; nt < NFEAT / 16; nt++) {
        const int ncol = nt * 16 + l15;
        v16h b0 = Bf[(nt * 2 + 0) * 32 + lane];
        v16h b1 = Bf[(nt * 2 + 1) * 32 + lane];
        v8f c = {};
        c = __builtin_amdgcn_wmma_f32_16x16x32_f16(false, a0, false, b0,
                                                   (short)0, c, false, false);
        c = __builtin_amdgcn_wmma_f32_16x16x32_f16(false, a1, false, b1,
                                                   (short)0, c, false, false);
        const float bv = bias[ncol];
#pragma unroll
        for (int r = 0; r < 8; r++) {
            int mo = mt * 16 + r + half * 8;
            float v = c[r] + bv;
            if (geluOut) v = gelu_f(v);
            out[mo * NFEAT + ncol] = v;
        }
    }
}

// ---------------------- per-row LayerNorm + GELU ---------------------------
// One wave per node, 2 features per lane, shfl_xor tree reduction (wave32).
__global__ void ln_gelu_kernel(const float* __restrict__ in,
                               const float* __restrict__ g,
                               const float* __restrict__ b,
                               float* __restrict__ out, int N) {
    const int lane = threadIdx.x & 31;
    const int wave = threadIdx.x >> 5;
    const int node = blockIdx.x * (blockDim.x >> 5) + wave;
    if (node >= N) return;
    float v0 = in[(long long)node * NFEAT + lane];
    float v1 = in[(long long)node * NFEAT + lane + 32];
    float s = v0 + v1;
    float q = v0 * v0 + v1 * v1;
#pragma unroll
    for (int msk = 16; msk > 0; msk >>= 1) {
        s += __shfl_xor(s, msk, 32);
        q += __shfl_xor(q, msk, 32);
    }
    float mean = s * (1.0f / 64.0f);
    float var  = q * (1.0f / 64.0f) - mean * mean;
    float iv   = rsqrtf(var + 1e-5f);
    float y0 = gelu_f((v0 - mean) * iv * g[lane] + b[lane]);
    float y1 = gelu_f((v1 - mean) * iv * g[lane + 32] + b[lane + 32]);
    out[(long long)node * NFEAT + lane]      = y0;
    out[(long long)node * NFEAT + lane + 32] = y1;
}

// ---------------------------------------------------------------------------

extern "C" void kernel_launch(void* const* d_in, const int* in_sizes, int n_in,
                              void* d_out, int out_size, void* d_ws, size_t ws_size,
                              hipStream_t stream) {
    const int N = in_sizes[0] / NFEAT;   // 50000 (multiple of 16)
    const int E = in_sizes[1] / 2;       // 800000

    const float* x       = (const float*)d_in[0];
    const int*   ei_phys = (const int*)d_in[1];
    const int*   ei_reg  = (const int*)d_in[2];

    struct Gin { const float *eps, *W1, *b1, *g1, *bt1, *W2, *b2; };
    Gin gin[2][2];
    const float *ln_g[2], *ln_b[2];
    int ii = 3;
    for (int l = 0; l < 2; l++) {
        for (int et = 0; et < 2; et++) {
            gin[l][et].eps = (const float*)d_in[ii++];
            gin[l][et].W1  = (const float*)d_in[ii++];
            gin[l][et].b1  = (const float*)d_in[ii++];
            gin[l][et].g1  = (const float*)d_in[ii++];
            gin[l][et].bt1 = (const float*)d_in[ii++];
            gin[l][et].W2  = (const float*)d_in[ii++];
            gin[l][et].b2  = (const float*)d_in[ii++];
        }
        ln_g[l] = (const float*)d_in[ii++];
        ln_b[l] = (const float*)d_in[ii++];
    }
    const float* hW1 = (const float*)d_in[ii++];
    const float* hb1 = (const float*)d_in[ii++];
    const float* hW2 = (const float*)d_in[ii++];
    const float* hb2 = (const float*)d_in[ii++];

    // ---- workspace layout (256B-aligned bump allocator) ----
    char* base = (char*)d_ws;
    size_t off = 0;
    auto alloc = [&](size_t bytes) -> char* {
        char* p = base + off;
        off = (off + bytes + 255) & ~(size_t)255;
        return p;
    };
    float* hA     = (float*)alloc((size_t)N * NFEAT * 4);
    float* hB     = (float*)alloc((size_t)N * NFEAT * 4);
    float* agg    = (float*)alloc((size_t)N * NFEAT * 4);
    float* z      = (float*)alloc((size_t)N * GH * 4);
    float* outsum = (float*)alloc((size_t)N * NFEAT * 4);
    float* stats  = (float*)alloc(2 * GH * 4);
    _Float16* w1h[2][2];
    _Float16* w2h[2][2];
    for (int l = 0; l < 2; l++)
        for (int et = 0; et < 2; et++) {
            w1h[l][et] = (_Float16*)alloc(NFEAT * GH * 2);
            w2h[l][et] = (_Float16*)alloc(GH * NFEAT * 2);
        }
    _Float16* hW1h = (_Float16*)alloc(NFEAT * NFEAT * 2);
    _Float16* hW2h = (_Float16*)alloc(NFEAT * NFEAT * 2);
    (void)ws_size; (void)n_in; (void)out_size;

    // ---- pack weights f32 -> f16 WMMA-fragment order ----
    for (int l = 0; l < 2; l++)
        for (int et = 0; et < 2; et++) {
            pack_b_kernel<<<(NFEAT * GH + 255) / 256, 256, 0, stream>>>(
                gin[l][et].W1, w1h[l][et], NFEAT, GH);
            pack_b_kernel<<<(GH * NFEAT + 255) / 256, 256, 0, stream>>>(
                gin[l][et].W2, w2h[l][et], GH, NFEAT);
        }
    pack_b_kernel<<<(NFEAT * NFEAT + 255) / 256, 256, 0, stream>>>(hW1, hW1h, NFEAT, NFEAT);
    pack_b_kernel<<<(NFEAT * NFEAT + 255) / 256, 256, 0, stream>>>(hW2, hW2h, NFEAT, NFEAT);

    const int mtiles     = N / 16;                 // 3125
    const int m_blocks   = (mtiles + 3) / 4;       // 4 waves (M-strips) / block
    const int agg_elems  = N * NFEAT;
    const int scat_total = E * 16;

    const float* cur = x;
    float* hbuf[2] = {hA, hB};

    for (int l = 0; l < 2; l++) {
        for (int et = 0; et < 2; et++) {
            const Gin& p = gin[l][et];
            const int* ei = et ? ei_reg : ei_phys;
            zero_kernel<<<(agg_elems + 255) / 256, 256, 0, stream>>>(agg, agg_elems);
            scatter_kernel<<<(scat_total + 255) / 256, 256, 0, stream>>>(
                cur, ei /*src*/, ei + E /*dst*/, agg, scat_total);
            gemm1_kernel<<<m_blocks, 128, 0, stream>>>(
                cur, agg, p.eps, (const v16h*)w1h[l][et], p.b1, z, N);
            colstats_kernel<<<GH, 256, 0, stream>>>(z, stats, N);
            gemm2_kernel<<<m_blocks, 128, 0, stream>>>(
                z, stats, p.g1, p.bt1, (const v16h*)w2h[l][et], p.b2, outsum,
                N, /*accum=*/et);
        }
        ln_gelu_kernel<<<(N + 7) / 8, 256, 0, stream>>>(
            outsum, ln_g[l], ln_b[l], hbuf[l], N);
        cur = hbuf[l];
    }

    // head: gelu(cur @ W1 + b1) @ W2 + b2   (cur == hB, temp == hA)
    gemm_plain_kernel<<<m_blocks, 128, 0, stream>>>(cur, (const v16h*)hW1h, hb1, hA, N, 1);
    gemm_plain_kernel<<<m_blocks, 128, 0, stream>>>(hA, (const v16h*)hW2h, hb2, (float*)d_out, N, 0);
}
